// RoPECrossAttentionLayer_29609504539063
// MI455X (gfx1250) — compile-verified, matrix operand order checked by reading, not run on hardware
//
#include <hip/hip_runtime.h>
#include <hip/hip_bf16.h>

// ---------------------------------------------------------------------------
// MI455X (gfx1250) transformer decoder layer: bf16 WMMA GEMMs + flash attention
// with Tensor Data Mover (TDM) LDS stage-in for non-transposed tiles.
// B=2, TQ=TKV=2048, D=512, H=8, HD=64, DFF=2048
// ---------------------------------------------------------------------------

#define DI __device__ __forceinline__

typedef __attribute__((ext_vector_type(16))) __bf16 v16bf;
typedef __attribute__((ext_vector_type(8)))  float  v8f;
typedef __attribute__((ext_vector_type(4)))  unsigned int v4u;
typedef __attribute__((ext_vector_type(8)))  int    v8i;
typedef __attribute__((ext_vector_type(4)))  int    v4i;

union BF16x16 {
    v16bf v;
    unsigned int   u32[8];
    unsigned short u16[16];
};

DI unsigned short f2bf(float f) {
    unsigned int u = __float_as_uint(f);
    unsigned int r = (u + 0x7FFFu + ((u >> 16) & 1u)) >> 16;   // RNE
    return (unsigned short)r;
}

// K index of the j-th bf16 pair in a 32-deep 16-bit fragment (ISA 7.12.2):
// lanes 0-15 hold K=0..7 then 16..23; lanes 16-31 hold K=8..15 then 24..31.
DI int kpair(int lane, int j) {
    int hi = (lane >> 4) * 8;
    return (j < 4) ? (hi + 2 * j) : (16 + hi + 2 * (j - 4));
}

#define WMMA_BF16(Af, Bf, Cv) \
    __builtin_amdgcn_wmma_f32_16x16x32_bf16(false, (Af), false, (Bf), (short)0, (Cv), false, false)

// ---------------------------------------------------------------------------
// Tensor Data Mover: DMA a 2D tile of bf16 (tileW elems x tileH rows, row
// stride strideElems) from global memory into LDS at ldsAddr, with 1-DWORD
// LDS padding every (1<<(padCode+1)) DWORDs (ISA 8.3/8.4 D# packing).
// Issue from a single wave; pairs with s_wait_tensorcnt + workgroup barrier.
// ---------------------------------------------------------------------------
#ifndef __has_builtin
#define __has_builtin(x) 0
#endif
#if __has_builtin(__builtin_amdgcn_tensor_load_to_lds) && \
    __has_builtin(__builtin_amdgcn_s_wait_tensorcnt)
#define HAVE_TDM 1
#else
#define HAVE_TDM 0
#endif

#if HAVE_TDM
DI void tdm_load_2d_bf16(unsigned ldsAddr, const void* gptr,
                         unsigned tileW, unsigned tileH,
                         unsigned strideElems, unsigned padCode) {
    unsigned long long ga = (unsigned long long)(size_t)gptr;
    v4u g0;
    g0.x = 1u;                                   // count=1, user load descriptor
    g0.y = ldsAddr;                              // D#.lds_addr (bytes)
    g0.z = (unsigned)(ga & 0xffffffffu);         // global_addr[31:0]
    g0.w = (unsigned)(ga >> 32) | (2u << 30);    // global_addr[56:32] | type=2
    const unsigned dim0 = 1u << 20, dim1 = 1u << 20;   // oversized (no OOB clip)
    v8i g1;
    g1[0] = (int)((1u << 16) | (1u << 20) | (padCode << 22)); // data_size=2B, pad_en, pad_interval, pad_amt=1dw
    g1[1] = (int)((dim0 & 0xffffu) << 16);                    // barrier addr 0 | dim0[15:0]
    g1[2] = (int)(((dim0 >> 16) & 0xffffu) | ((dim1 & 0xffffu) << 16));
    g1[3] = (int)(((dim1 >> 16) & 0xffffu) | (tileW << 16));  // dim1 hi | tile_dim0
    g1[4] = (int)(tileH & 0xffffu);                           // tile_dim1 | tile_dim2=0
    g1[5] = (int)strideElems;                                 // tensor_dim0_stride[31:0]
    g1[6] = 0;                                                // stride0 hi | stride1 lo
    g1[7] = 0;
    v4i gz = {0, 0, 0, 0};
#if __clang_major__ >= 23
    v8i gz8 = {0, 0, 0, 0, 0, 0, 0, 0};
    __builtin_amdgcn_tensor_load_to_lds(g0, g1, gz, gz, gz8, 0);
#else
    __builtin_amdgcn_tensor_load_to_lds(g0, g1, gz, gz, 0);
#endif
}
#endif

// ---------------------------------------------------------------------------
// fp32 -> bf16 conversion
// ---------------------------------------------------------------------------
__global__ __launch_bounds__(256)
void cvt_bf16_kernel(const float* __restrict__ in, unsigned short* __restrict__ out, int n) {
    int i = blockIdx.x * 256 + threadIdx.x;
    if (i < n) out[i] = f2bf(in[i]);
}

// ---------------------------------------------------------------------------
// LayerNorm over D=512, one row per block, bf16 output for WMMA consumption.
// ---------------------------------------------------------------------------
__global__ __launch_bounds__(256)
void layernorm_bf16_kernel(const float* __restrict__ X, const float* __restrict__ g,
                           const float* __restrict__ bta, unsigned short* __restrict__ Y) {
    __shared__ float red[8];
    const int tid = threadIdx.x;
    const size_t row = blockIdx.x;
    const float* x = X + row * 512;
    float a = x[tid], c = x[tid + 256];

    float s = a + c;
    #pragma unroll
    for (int off = 1; off < 32; off <<= 1) s += __shfl_xor(s, off, 32);
    if ((tid & 31) == 0) red[tid >> 5] = s;
    __syncthreads();
    float tot = 0.f;
    #pragma unroll
    for (int i = 0; i < 8; ++i) tot += red[i];
    const float mean = tot * (1.0f / 512.0f);
    __syncthreads();

    const float da = a - mean, dc = c - mean;
    float vs = da * da + dc * dc;
    #pragma unroll
    for (int off = 1; off < 32; off <<= 1) vs += __shfl_xor(vs, off, 32);
    if ((tid & 31) == 0) red[tid >> 5] = vs;
    __syncthreads();
    float vtot = 0.f;
    #pragma unroll
    for (int i = 0; i < 8; ++i) vtot += red[i];
    const float rstd = 1.0f / sqrtf(vtot * (1.0f / 512.0f) + 1e-5f);

    Y[row * 512 + tid]       = f2bf(da * rstd * g[tid]       + bta[tid]);
    Y[row * 512 + tid + 256] = f2bf(dc * rstd * g[tid + 256] + bta[tid + 256]);
}

// ---------------------------------------------------------------------------
// RoPE on fp32 activations, bf16 output. Layout [bt][h*64+d], HD=64, half=32.
// ---------------------------------------------------------------------------
__global__ __launch_bounds__(256)
void rope_bf16_kernel(const float* __restrict__ X, const int* __restrict__ pos,
                      unsigned short* __restrict__ Y, int nrows) {
    int id = blockIdx.x * 256 + threadIdx.x;
    int bt = id >> 8;
    if (bt >= nrows) return;
    int p = id & 255;
    int hh = p >> 5, i = p & 31;
    int d = hh * 64 + i;
    float inv = expf(-(float)(2 * i) * (1.0f / 64.0f) * 6.907755278982137f); // ln(1000)
    float f = (float)pos[bt] * inv;
    float cs = cosf(f), sn = sinf(f);
    size_t base = (size_t)bt * 512;
    float x0 = X[base + d], x1 = X[base + d + 32];
    Y[base + d]      = f2bf(x0 * cs - x1 * sn);
    Y[base + d + 32] = f2bf(x1 * cs + x0 * sn);
}

// ---------------------------------------------------------------------------
// Generic bf16 GEMM: C[M,N] = A[M,K] @ W[K,N] + bias (+ GELU) (+ residual).
// Block tile 128x64, 8 waves, each wave 32x32 via 2x2 v_wmma_f32_16x16x32_bf16.
// A tile staged by TDM (row-major, HW-padded); W tile scatter-transposed.
// ---------------------------------------------------------------------------
__global__ __launch_bounds__(256)
void gemm_bf16_kernel(const unsigned short* __restrict__ A,
                      const unsigned short* __restrict__ W,
                      const float* __restrict__ bias,
                      const float* __restrict__ resid,
                      float* __restrict__ outF,
                      unsigned short* __restrict__ outB,
                      int M, int N, int K, int gelu) {
    __shared__ unsigned short sA[128 * 34];  // [row][k]: 64B data + 4B pad per row
    __shared__ unsigned short sB[64 * 34];   // [col][k], transposed

    const int tid = threadIdx.x;
    const int lane = tid & 31, w = tid >> 5;
    const int m0 = blockIdx.y * 128, n0 = blockIdx.x * 64;
    const int wm = (w >> 1) * 32, wn = (w & 1) * 32;
    const int colL = lane & 15, rowHi = (lane >> 4) * 8;

    v8f acc[2][2];
    #pragma unroll
    for (int i = 0; i < 2; ++i)
        #pragma unroll
        for (int j = 0; j < 2; ++j)
            #pragma unroll
            for (int e = 0; e < 8; ++e) acc[i][j][e] = 0.f;

    for (int kk = 0; kk < K; kk += 32) {
#if HAVE_TDM
        if (tid == 0) {  // TDM: 128 rows x 32 bf16, pad 1 DWORD per 16 DWORDs
            tdm_load_2d_bf16((unsigned)(unsigned long long)(size_t)&sA[0],
                             A + (size_t)m0 * K + kk, 32u, 128u, (unsigned)K, 3u);
            __builtin_amdgcn_s_wait_tensorcnt(0);
        }
#else
        {   // A tile 128x32: thread loads 16 bf16 (8 dwords), row-major into LDS
            int r = tid >> 1, c0 = (tid & 1) * 16;
            const unsigned int* ga = (const unsigned int*)(A + (size_t)(m0 + r) * K + kk + c0);
            unsigned int* la = (unsigned int*)(sA + r * 34 + c0);
            #pragma unroll
            for (int i = 0; i < 8; ++i) la[i] = ga[i];
        }
#endif
        {   // W tile 32x64: load row-major, scatter transposed [col][k]
            int r = tid >> 3, c0 = (tid & 7) * 8;
            const unsigned int* gb = (const unsigned int*)(W + (size_t)(kk + r) * N + n0 + c0);
            #pragma unroll
            for (int i = 0; i < 4; ++i) {
                unsigned int u = gb[i];
                sB[(c0 + 2 * i) * 34 + r]     = (unsigned short)(u & 0xffffu);
                sB[(c0 + 2 * i + 1) * 34 + r] = (unsigned short)(u >> 16);
            }
            if (kk + 32 < K) __builtin_prefetch(gb + (size_t)32 * (N >> 1), 0, 1);
        }
        __syncthreads();

        BF16x16 a0, a1, b0, b1;
        #pragma unroll
        for (int j = 0; j < 8; ++j) {
            int k = kpair(lane, j);
            a0.u32[j] = *(const unsigned int*)(sA + (wm + colL) * 34 + k);
            a1.u32[j] = *(const unsigned int*)(sA + (wm + 16 + colL) * 34 + k);
            b0.u32[j] = *(const unsigned int*)(sB + (wn + colL) * 34 + k);
            b1.u32[j] = *(const unsigned int*)(sB + (wn + 16 + colL) * 34 + k);
        }
        acc[0][0] = WMMA_BF16(a0.v, b0.v, acc[0][0]);
        acc[0][1] = WMMA_BF16(a0.v, b1.v, acc[0][1]);
        acc[1][0] = WMMA_BF16(a1.v, b0.v, acc[1][0]);
        acc[1][1] = WMMA_BF16(a1.v, b1.v, acc[1][1]);
        __syncthreads();
    }

    #pragma unroll
    for (int i = 0; i < 2; ++i)
        #pragma unroll
        for (int jt = 0; jt < 2; ++jt)
            #pragma unroll
            for (int e = 0; e < 8; ++e) {
                size_t row = (size_t)(m0 + wm + i * 16 + e + rowHi);
                int col = n0 + wn + jt * 16 + colL;
                float vv = acc[i][jt][e] + bias[col];
                if (gelu) vv = 0.5f * vv * (1.0f + erff(vv * 0.70710678118654752f));
                if (resid) vv += resid[row * N + col];
                if (outF) outF[row * N + col] = vv;
                if (outB) outB[row * N + col] = f2bf(vv);
            }
}

// ---------------------------------------------------------------------------
// Flash attention (bf16 QKV, fp32 softmax). Block = 128 threads = 4 waves,
// each wave owns 16 query rows; KV streamed in 64-wide chunks through LDS.
// K tile staged by TDM (row-major, HW-padded); V tile scatter-transposed.
// ---------------------------------------------------------------------------
__global__ __launch_bounds__(128)
void flash_attn_kernel(const unsigned short* __restrict__ Q,
                       const unsigned short* __restrict__ Kb,
                       const unsigned short* __restrict__ Vb,
                       const unsigned char* __restrict__ mask,
                       unsigned short* __restrict__ O,
                       int TQn, int TKVn) {
    __shared__ unsigned short sK[64 * 66];       // [kv][hd]: 128B data + 4B pad
    __shared__ unsigned short sV[64 * 66];       // [hd][kv] (transposed)
    __shared__ unsigned short sP[4 * 16 * 66];   // per-wave P staging [qrow][kv]
    __shared__ float sM[64];

    const int tid = threadIdx.x, lane = tid & 31, w = tid >> 5;
    const int h = blockIdx.y, b = blockIdx.z;
    const int q0 = blockIdx.x * 64 + w * 16;
    const int colL = lane & 15, rowHi = (lane >> 4) * 8;
    const float SC = 0.125f;  // 1/sqrt(HD)

    // Q fragments (A-layout, two 32-deep halves of HD=64), kept in registers
    BF16x16 qf[2];
    {
        size_t base = ((size_t)(b * TQn + q0 + colL) * 512) + h * 64;
        #pragma unroll
        for (int kh = 0; kh < 2; ++kh)
            #pragma unroll
            for (int j = 0; j < 8; ++j)
                qf[kh].u32[j] = *(const unsigned int*)(Q + base + kh * 32 + kpair(lane, j));
    }

    float m8[8], l8[8];
    v8f o[4];
    #pragma unroll
    for (int e = 0; e < 8; ++e) { m8[e] = -1e30f; l8[e] = 0.f; }
    #pragma unroll
    for (int nt = 0; nt < 4; ++nt)
        #pragma unroll
        for (int e = 0; e < 8; ++e) o[nt][e] = 0.f;

    for (int k0 = 0; k0 < TKVn; k0 += 64) {
#if HAVE_TDM
        if (tid == 0) {  // TDM: 64 rows x 64 bf16, pad 1 DWORD per 32 DWORDs
            tdm_load_2d_bf16((unsigned)(unsigned long long)(size_t)&sK[0],
                             Kb + ((size_t)(b * TKVn + k0) * 512) + h * 64,
                             64u, 64u, 512u, 4u);
            __builtin_amdgcn_s_wait_tensorcnt(0);
        }
#endif
        {   // stage V (transposed); mask row in fp32 additive form
            int r = tid >> 1, c0 = (tid & 1) * 32;
            size_t gbase = ((size_t)(b * TKVn + k0 + r) * 512) + h * 64 + c0;
            const unsigned int* gv = (const unsigned int*)(Vb + gbase);
#if !HAVE_TDM
            const unsigned int* gk = (const unsigned int*)(Kb + gbase);
#endif
            #pragma unroll
            for (int i = 0; i < 16; ++i) {
#if !HAVE_TDM
                *(unsigned int*)(sK + r * 66 + c0 + 2 * i) = gk[i];
#endif
                unsigned int u = gv[i];
                sV[(c0 + 2 * i) * 66 + r]     = (unsigned short)(u & 0xffffu);
                sV[(c0 + 2 * i + 1) * 66 + r] = (unsigned short)(u >> 16);
            }
            if (tid < 64)
                sM[tid] = (mask && mask[(size_t)b * TKVn + k0 + tid]) ? -1e30f : 0.0f;
        }
        __syncthreads();

        // S = Q @ K^T  (4 column tiles of 16 keys, each 2 WMMAs over HD)
        v8f s[4];
        #pragma unroll
        for (int nt = 0; nt < 4; ++nt) {
            BF16x16 b0, b1;
            int kv = nt * 16 + colL;
            #pragma unroll
            for (int j = 0; j < 8; ++j) {
                int k = kpair(lane, j);
                b0.u32[j] = *(const unsigned int*)(sK + kv * 66 + k);
                b1.u32[j] = *(const unsigned int*)(sK + kv * 66 + 32 + k);
            }
            v8f a;
            #pragma unroll
            for (int e = 0; e < 8; ++e) a[e] = 0.f;
            a = WMMA_BF16(qf[0].v, b0.v, a);
            a = WMMA_BF16(qf[1].v, b1.v, a);
            float ma = sM[kv];
            #pragma unroll
            for (int e = 0; e < 8; ++e) s[nt][e] = a[e] * SC + ma;
        }

        // online softmax (rows live across 16 lanes of a half-wave)
        #pragma unroll
        for (int e = 0; e < 8; ++e) {
            float t = fmaxf(fmaxf(s[0][e], s[1][e]), fmaxf(s[2][e], s[3][e]));
            t = fmaxf(t, __shfl_xor(t, 1, 32));
            t = fmaxf(t, __shfl_xor(t, 2, 32));
            t = fmaxf(t, __shfl_xor(t, 4, 32));
            t = fmaxf(t, __shfl_xor(t, 8, 32));
            float nm = fmaxf(m8[e], t);
            float corr = expf(m8[e] - nm);
            m8[e] = nm;
            l8[e] *= corr;
            float ps = 0.f;
            #pragma unroll
            for (int nt = 0; nt < 4; ++nt) {
                float p = expf(s[nt][e] - nm);
                ps += p;
                sP[(w * 16 + e + rowHi) * 66 + nt * 16 + colL] = f2bf(p);
                o[nt][e] *= corr;
            }
            ps += __shfl_xor(ps, 1, 32);
            ps += __shfl_xor(ps, 2, 32);
            ps += __shfl_xor(ps, 4, 32);
            ps += __shfl_xor(ps, 8, 32);
            l8[e] += ps;
        }
        __syncthreads();

        // O += P @ V
        BF16x16 pf[2];
        {
            int base = (w * 16 + colL) * 66;
            #pragma unroll
            for (int kh = 0; kh < 2; ++kh)
                #pragma unroll
                for (int j = 0; j < 8; ++j)
                    pf[kh].u32[j] = *(const unsigned int*)(sP + base + kh * 32 + kpair(lane, j));
        }
        #pragma unroll
        for (int nt = 0; nt < 4; ++nt) {
            BF16x16 v0, v1;
            int hd = nt * 16 + colL;
            #pragma unroll
            for (int j = 0; j < 8; ++j) {
                int k = kpair(lane, j);
                v0.u32[j] = *(const unsigned int*)(sV + hd * 66 + k);
                v1.u32[j] = *(const unsigned int*)(sV + hd * 66 + 32 + k);
            }
            o[nt] = WMMA_BF16(pf[0].v, v0.v, o[nt]);
            o[nt] = WMMA_BF16(pf[1].v, v1.v, o[nt]);
        }
        __syncthreads();
    }

    #pragma unroll
    for (int e = 0; e < 8; ++e) {
        float inv = 1.0f / l8[e];
        size_t row = (size_t)(b * TQn + q0 + e + rowHi);
        #pragma unroll
        for (int nt = 0; nt < 4; ++nt)
            O[row * 512 + h * 64 + nt * 16 + colL] = f2bf(o[nt][e] * inv);
    }
}

// ---------------------------------------------------------------------------
// Host-side orchestration
// ---------------------------------------------------------------------------
extern "C" void kernel_launch(void* const* d_in, const int* in_sizes, int n_in,
                              void* d_out, int out_size, void* d_ws, size_t ws_size,
                              hipStream_t stream) {
    (void)in_sizes; (void)n_in; (void)out_size; (void)ws_size;

    const int BATCH = 2, TQ = 2048, TKV = 2048, D = 512, DFF = 2048;
    const int ROWS = BATCH * TQ;          // 4096
    const int NACT = ROWS * D;            // 2,097,152

    const float* x    = (const float*)d_in[0];
    const float* enc  = (const float*)d_in[1];
    const int*   qpos = (const int*)d_in[2];
    const int*   kpos = (const int*)d_in[3];
    const unsigned char* msk = (const unsigned char*)d_in[4];

    const float* wf[10]; const float* bf[10];
    for (int i = 0; i < 10; ++i) { wf[i] = (const float*)d_in[5 + 2 * i]; bf[i] = (const float*)d_in[6 + 2 * i]; }
    const float* ln1g = (const float*)d_in[25]; const float* ln1b = (const float*)d_in[26];
    const float* ln2g = (const float*)d_in[27]; const float* ln2b = (const float*)d_in[28];
    const float* ln3g = (const float*)d_in[29]; const float* ln3b = (const float*)d_in[30];

    size_t off = 0;
    auto alloc = [&](size_t bytes) -> void* {
        void* p = (void*)((char*)d_ws + off);
        off = (off + bytes + 255) & ~(size_t)255;
        return p;
    };

    const int wElems[10] = { D*D, D*D, D*D, D*D, D*D, D*D, D*D, D*D, D*DFF, DFF*D };
    unsigned short* wbf[10];
    for (int i = 0; i < 10; ++i) wbf[i] = (unsigned short*)alloc((size_t)wElems[i] * 2);

    unsigned short* xnbf   = (unsigned short*)alloc((size_t)NACT * 2);
    float*          qf32   = (float*)alloc((size_t)NACT * 4);
    float*          kf32   = (float*)alloc((size_t)NACT * 4);
    unsigned short* qbf    = (unsigned short*)alloc((size_t)NACT * 2);
    unsigned short* kbf    = (unsigned short*)alloc((size_t)NACT * 2);
    unsigned short* vbf    = (unsigned short*)alloc((size_t)NACT * 2);
    unsigned short* attnbf = (unsigned short*)alloc((size_t)NACT * 2);
    float*          x1     = (float*)alloc((size_t)NACT * 4);
    float*          x2     = (float*)alloc((size_t)NACT * 4);
    unsigned short* encbf  = (unsigned short*)alloc((size_t)NACT * 2);
    unsigned short* hbf    = (unsigned short*)alloc((size_t)ROWS * DFF * 2);

    // Convert weights + encoder activations to bf16 once.
    for (int i = 0; i < 10; ++i)
        cvt_bf16_kernel<<<(wElems[i] + 255) / 256, 256, 0, stream>>>(wf[i], wbf[i], wElems[i]);
    cvt_bf16_kernel<<<(NACT + 255) / 256, 256, 0, stream>>>(enc, encbf, NACT);

    dim3 gGemm(D / 64, ROWS / 128);        // (8, 32) for N=512
    dim3 gGemmF1(DFF / 64, ROWS / 128);    // (32, 32) for N=2048
    dim3 gAttn(TQ / 64, 8, BATCH);         // (32, 8, 2)

    // ---- self-attention ----
    layernorm_bf16_kernel<<<ROWS, 256, 0, stream>>>(x, ln1g, ln1b, xnbf);
    gemm_bf16_kernel<<<gGemm, 256, 0, stream>>>(xnbf, wbf[0], bf[0], nullptr, qf32, nullptr, ROWS, D, D, 0);
    gemm_bf16_kernel<<<gGemm, 256, 0, stream>>>(xnbf, wbf[1], bf[1], nullptr, kf32, nullptr, ROWS, D, D, 0);
    gemm_bf16_kernel<<<gGemm, 256, 0, stream>>>(xnbf, wbf[2], bf[2], nullptr, nullptr, vbf, ROWS, D, D, 0);
    rope_bf16_kernel<<<ROWS, 256, 0, stream>>>(qf32, qpos, qbf, ROWS);
    rope_bf16_kernel<<<ROWS, 256, 0, stream>>>(kf32, qpos, kbf, ROWS);
    flash_attn_kernel<<<gAttn, 128, 0, stream>>>(qbf, kbf, vbf, nullptr, attnbf, TQ, TQ);
    gemm_bf16_kernel<<<gGemm, 256, 0, stream>>>(attnbf, wbf[3], bf[3], x, x1, nullptr, ROWS, D, D, 0);

    // ---- cross-attention ----
    layernorm_bf16_kernel<<<ROWS, 256, 0, stream>>>(x1, ln2g, ln2b, xnbf);
    gemm_bf16_kernel<<<gGemm, 256, 0, stream>>>(xnbf, wbf[4], bf[4], nullptr, qf32, nullptr, ROWS, D, D, 0);
    gemm_bf16_kernel<<<gGemm, 256, 0, stream>>>(encbf, wbf[5], bf[5], nullptr, kf32, nullptr, ROWS, D, D, 0);
    gemm_bf16_kernel<<<gGemm, 256, 0, stream>>>(encbf, wbf[6], bf[6], nullptr, nullptr, vbf, ROWS, D, D, 0);
    rope_bf16_kernel<<<ROWS, 256, 0, stream>>>(qf32, qpos, qbf, ROWS);
    rope_bf16_kernel<<<ROWS, 256, 0, stream>>>(kf32, kpos, kbf, ROWS);
    flash_attn_kernel<<<gAttn, 128, 0, stream>>>(qbf, kbf, vbf, msk, attnbf, TQ, TKV);
    gemm_bf16_kernel<<<gGemm, 256, 0, stream>>>(attnbf, wbf[7], bf[7], x1, x2, nullptr, ROWS, D, D, 0);

    // ---- FFN ----
    layernorm_bf16_kernel<<<ROWS, 256, 0, stream>>>(x2, ln3g, ln3b, xnbf);
    gemm_bf16_kernel<<<gGemmF1, 256, 0, stream>>>(xnbf, wbf[8], bf[8], nullptr, nullptr, hbf, ROWS, DFF, D, 1);
    gemm_bf16_kernel<<<gGemm, 256, 0, stream>>>(hbf, wbf[9], bf[9], x2, (float*)d_out, nullptr, ROWS, D, DFF, 0);
}
